// Transform_17927193494080
// MI455X (gfx1250) — matrix operand order repaired and checked by reading, not compiled, not run on hardware
//
#include <hip/hip_runtime.h>
#include <hip/hip_bf16.h>
#include <stdint.h>

// ---------------------------------------------------------------------------
// Fused transformer block for MI455X (gfx1250, wave32, WMMA).
// ~150 GFLOP vs ~33MB HBM traffic -> matrix-engine bound. All GEMMs go
// through v_wmma_f32_16x16x32_f16 (f32 accumulate). Weights are converted to
// f16 ONCE (k_cvt4) so the GEMM inner loops are pure global_load_b128 + wmma
// (no v_cvt in the hot path, half the weight L2 traffic). V is stored
// transposed so the P*V B-operand is a contiguous load. The C->A relayout of
// attention probabilities uses a wave-private LDS tile + s_wait_dscnt.
// ---------------------------------------------------------------------------

#define DI __device__ __forceinline__

typedef __attribute__((ext_vector_type(16))) _Float16 v16h;
typedef __attribute__((ext_vector_type(8)))  float    v8f;

struct __align__(16) Pack32 { uint4 a, b; };   // 32 bytes
struct H16 { _Float16 h[16]; };                // 32 bytes
struct __align__(16) H8 { _Float16 h[8]; };    // 16 bytes
struct __align__(8)  H4 { _Float16 h[4]; };    // 8 bytes

static constexpr int   BB      = 4;
static constexpr int   SS      = 4096;
static constexpr int   DD      = 256;
static constexpr float LN_EPSF = 1e-5f;
static constexpr float INV_SQD = 0.0625f;                 // 1/sqrt(256)
static constexpr float MASKEDV = -32767.0f * 0.0625f;     // mask applied pre-scale

DI v8f vzero() { v8f z = {0.f,0.f,0.f,0.f,0.f,0.f,0.f,0.f}; return z; }
DI v8f vsplat(float f) { v8f z = {f,f,f,f,f,f,f,f}; return z; }

DI v8f wmma16(v16h a, v16h b, v8f c) {
  // (neg_a, A, neg_b, B, c_mod, C, reuse_a, reuse_b)
  return __builtin_amdgcn_wmma_f32_16x16x32_f16(false, a, false, b, (short)0, c,
                                                false, false);
}

// A-frag (16x32 f16): lane<16 holds K = p0[0..7] then p1[0..7] (K=8..15);
// lane>=16 gets +16 via the caller's pointer offsets.
DI v16h ld_pair(const _Float16* p0, const _Float16* p1) {
  Pack32 t;
  t.a = *reinterpret_cast<const uint4*>(p0);
  t.b = *reinterpret_cast<const uint4*>(p1);
  return __builtin_bit_cast(v16h, t);
}
// B-frag (32x16 f16): lane's column = lane&15, element e -> K = 16*(lane>>4)+e,
// contiguous 16 f16 in memory.
DI v16h ld_row16(const _Float16* p) { return ld_pair(p, p + 8); }

DI v16h cvt_pair(const float* p0, const float* p1) {
  H16 t;
#pragma unroll
  for (int i = 0; i < 8; ++i) { t.h[i] = (_Float16)p0[i]; t.h[8+i] = (_Float16)p1[i]; }
  return __builtin_bit_cast(v16h, t);
}

// LDS traffic here is wave-private; LDS is in-order per wave so a wave-local
// DScnt wait is the only fence needed (no workgroup barrier in divergent loops).
DI void wave_ds_fence() { asm volatile("s_wait_dscnt 0" ::: "memory"); }

// ---------------------------------------------------------------------------
// Kernel 0: one-shot f32 -> f16 weight conversion (640KB total, L2-resident).
// ---------------------------------------------------------------------------
__global__ __launch_bounds__(256) void k_cvt4(const float4* __restrict__ s,
                                              _Float16* __restrict__ d) {
  const int i = blockIdx.x * 256 + threadIdx.x;
  const float4 v = s[i];
  H4 p;
  p.h[0] = (_Float16)v.x; p.h[1] = (_Float16)v.y;
  p.h[2] = (_Float16)v.z; p.h[3] = (_Float16)v.w;
  *reinterpret_cast<H4*>(d + (size_t)i * 4) = p;
}

// ---------------------------------------------------------------------------
// One projection: y = x@W^T + b. TRANSP=false -> row-major [B,S,D] store;
// TRANSP=true -> transposed [B,D,S] store (single 16B store per tile).
// W16 comes straight from a kernel argument so loads stay on the GLOBAL path.
// ---------------------------------------------------------------------------
template <bool TRANSP>
DI void proj_one(const v16h af[8], const _Float16* __restrict__ W16,
                 const float* __restrict__ bias, _Float16* __restrict__ out,
                 int b, int qb, int half, int nl) {
  for (int nt = 0; nt < 16; ++nt) {
    const int col = nt * 16 + nl;
    v8f c = vsplat(bias[col]);   // bias folded into the accumulator init
#pragma unroll
    for (int ks = 0; ks < 8; ++ks) {
      // y = x @ W^T: B[k][n] = W[n][k] -> contiguous run of W row `col`.
      c = wmma16(af[ks], ld_row16(W16 + (size_t)col * DD + ks * 32 + half * 16), c);
    }
    if (!TRANSP) {
#pragma unroll
      for (int r = 0; r < 8; ++r)
        out[((size_t)b * SS + qb + half * 8 + r) * DD + col] = (_Float16)c[r];
    } else {
      H8 pk;
#pragma unroll
      for (int r = 0; r < 8; ++r) pk.h[r] = (_Float16)c[r];
      *reinterpret_cast<H8*>(out + ((size_t)b * DD + col) * SS + qb + half * 8) = pk;
    }
  }
}

// ---------------------------------------------------------------------------
// Kernel 1: Q,K = f16 row-major [B,S,D]; V stored transposed Vt = f16 [B,D,S].
// ---------------------------------------------------------------------------
__global__ __launch_bounds__(128) void k_qkv(
    const float* __restrict__ x,
    const _Float16* __restrict__ Wq16, const float* __restrict__ bq,
    const _Float16* __restrict__ Wk16, const float* __restrict__ bk,
    const _Float16* __restrict__ Wv16, const float* __restrict__ bv,
    _Float16* __restrict__ Q, _Float16* __restrict__ K,
    _Float16* __restrict__ Vt) {
  const int lane = threadIdx.x & 31;
  const int wid  = threadIdx.x >> 5;
  const int gw   = blockIdx.x * 4 + wid;
  const int b    = gw / (SS / 16);
  const int qb   = (gw % (SS / 16)) * 16;
  const int half = lane >> 4;
  const int nl   = lane & 15;

  // x row tile -> 8 A-fragments (K-steps of 32 over D=256), f32 -> f16 once.
  const float* xrow = x + ((size_t)b * SS + qb + nl) * DD;
  v16h af[8];
#pragma unroll
  for (int ks = 0; ks < 8; ++ks)
    af[ks] = cvt_pair(xrow + ks * 32 + half * 16,
                      xrow + ks * 32 + half * 16 + 8);

  proj_one<false>(af, Wq16, bq, Q,  b, qb, half, nl);
  proj_one<false>(af, Wk16, bk, K,  b, qb, half, nl);
  proj_one<true >(af, Wv16, bv, Vt, b, qb, half, nl);
}

// ---------------------------------------------------------------------------
// Kernel 2: causal flash attention + residual + layernorm -> f16 [B,S,D].
// ---------------------------------------------------------------------------
__global__ __launch_bounds__(128) void k_attn(
    const _Float16* __restrict__ Q, const _Float16* __restrict__ K,
    const _Float16* __restrict__ Vt, const float* __restrict__ x,
    const float* __restrict__ g_ln, const float* __restrict__ be_ln,
    _Float16* __restrict__ outln) {
  __shared__ _Float16 shp[4][16 * 40];   // per-wave 16x32 P tile, padded rows

  const int lane = threadIdx.x & 31;
  const int wid  = threadIdx.x >> 5;
  const int gw   = blockIdx.x * 4 + wid;
  const int b    = gw / (SS / 16);
  const int qb   = (gw % (SS / 16)) * 16;
  const int half = lane >> 4;
  const int nl   = lane & 15;
  _Float16* pbuf = &shp[wid][0];

  const _Float16* qrow = Q + ((size_t)b * SS + qb + nl) * DD;
  v16h qf[8];
#pragma unroll
  for (int ks = 0; ks < 8; ++ks)
    qf[ks] = ld_pair(qrow + ks * 32 + half * 16,
                     qrow + ks * 32 + half * 16 + 8);

  v8f o[16];
#pragma unroll
  for (int t = 0; t < 16; ++t) o[t] = vzero();
  float mi[8], li[8];
#pragma unroll
  for (int r = 0; r < 8; ++r) { mi[r] = -INFINITY; li[r] = 0.f; }

  const int nkb = (qb + 16 + 31) >> 5;   // causal: skip blocks above diagonal
  for (int kb = 0; kb < nkb; ++kb) {
    const int k0 = kb * 32;
    // Prefetch next key block while this block's WMMA chain executes.
    if (kb + 1 < nkb) {
      __builtin_prefetch(K  + ((size_t)b * SS + k0 + 32 + nl) * DD, 0, 1);
      __builtin_prefetch(Vt + ((size_t)b * DD + nl) * SS + k0 + 32, 0, 1);
    }
    // --- S = Q K^T, 16 q-rows x 32 keys, f32 accumulate -------------------
    v8f s0 = vzero(), s1 = vzero();
#pragma unroll
    for (int ks = 0; ks < 8; ++ks) {
      const _Float16* kr0 = K + ((size_t)b * SS + k0 + nl) * DD + ks * 32 + half * 16;
      const _Float16* kr1 = K + ((size_t)b * SS + k0 + 16 + nl) * DD + ks * 32 + half * 16;
      s0 = wmma16(qf[ks], ld_row16(kr0), s0);
      s1 = wmma16(qf[ks], ld_row16(kr1), s1);
    }
    // --- mask (pre-scale, faithful) + 1/sqrt(D) ---------------------------
    float sc0[8], sc1[8];
#pragma unroll
    for (int r = 0; r < 8; ++r) {
      const int row = qb + half * 8 + r;
      sc0[r] = (k0 + nl      <= row) ? s0[r] * INV_SQD : MASKEDV;
      sc1[r] = (k0 + 16 + nl <= row) ? s1[r] * INV_SQD : MASKEDV;
    }
    // --- online softmax; row stats live in 16-lane half-groups ------------
    float mn[8], al[8];
#pragma unroll
    for (int r = 0; r < 8; ++r) {
      float rm = fmaxf(sc0[r], sc1[r]);
#pragma unroll
      for (int m = 1; m < 16; m <<= 1) rm = fmaxf(rm, __shfl_xor(rm, m, 32));
      mn[r] = fmaxf(mi[r], rm);
      al[r] = __expf(mi[r] - mn[r]);
      mi[r] = mn[r];
    }
    float p0[8], p1[8];
#pragma unroll
    for (int r = 0; r < 8; ++r) {
      p0[r] = __expf(sc0[r] - mn[r]);
      p1[r] = __expf(sc1[r] - mn[r]);
      float ps = p0[r] + p1[r];
#pragma unroll
      for (int m = 1; m < 16; m <<= 1) ps += __shfl_xor(ps, m, 32);
      li[r] = li[r] * al[r] + ps;
    }
#pragma unroll
    for (int t = 0; t < 16; ++t)
#pragma unroll
      for (int r = 0; r < 8; ++r) o[t][r] = o[t][r] * al[r];
    // --- P: C-layout -> A-layout via wave-private LDS tile ----------------
#pragma unroll
    for (int r = 0; r < 8; ++r) {
      pbuf[(half * 8 + r) * 40 + nl]      = (_Float16)p0[r];
      pbuf[(half * 8 + r) * 40 + 16 + nl] = (_Float16)p1[r];
    }
    wave_ds_fence();
    v16h pf = ld_pair(&pbuf[nl * 40 + half * 16], &pbuf[nl * 40 + half * 16 + 8]);
    // --- O += P Vt: Vt rows make B-frags contiguous -----------------------
#pragma unroll
    for (int dt = 0; dt < 16; ++dt) {
      const _Float16* vr = Vt + ((size_t)b * DD + dt * 16 + nl) * SS + k0 + half * 16;
      o[dt] = wmma16(pf, ld_row16(vr), o[dt]);
    }
  }

  // --- normalize, residual, layernorm ------------------------------------
  float inv[8];
#pragma unroll
  for (int r = 0; r < 8; ++r) inv[r] = 1.0f / li[r];
  float gv[16], bv[16];
#pragma unroll
  for (int t = 0; t < 16; ++t) { gv[t] = g_ln[t * 16 + nl]; bv[t] = be_ln[t * 16 + nl]; }

  float sum[8], sq[8];
#pragma unroll
  for (int r = 0; r < 8; ++r) { sum[r] = 0.f; sq[r] = 0.f; }
#pragma unroll
  for (int t = 0; t < 16; ++t)
#pragma unroll
    for (int r = 0; r < 8; ++r) {
      float v = o[t][r] * inv[r] +
                x[((size_t)b * SS + qb + half * 8 + r) * DD + t * 16 + nl];
      o[t][r] = v;
      sum[r] += v; sq[r] += v * v;
    }
#pragma unroll
  for (int r = 0; r < 8; ++r) {
#pragma unroll
    for (int m = 1; m < 16; m <<= 1) {
      sum[r] += __shfl_xor(sum[r], m, 32);
      sq[r]  += __shfl_xor(sq[r],  m, 32);
    }
  }
  float mu[8], rs[8];
#pragma unroll
  for (int r = 0; r < 8; ++r) {
    mu[r] = sum[r] * (1.0f / DD);
    float var = sq[r] * (1.0f / DD) - mu[r] * mu[r];
    rs[r] = rsqrtf(var + LN_EPSF);
  }
#pragma unroll
  for (int t = 0; t < 16; ++t)
#pragma unroll
    for (int r = 0; r < 8; ++r)
      outln[((size_t)b * SS + qb + half * 8 + r) * DD + t * 16 + nl] =
          (_Float16)((o[t][r] - mu[r]) * rs[r] * gv[t] + bv[t]);
}

// ---------------------------------------------------------------------------
// Kernel 3: FFN (relu(h@W1^T+b1) @ W2^T + b2) + layernorm -> f32 output.
// ---------------------------------------------------------------------------
__global__ __launch_bounds__(128) void k_ffn(
    const _Float16* __restrict__ H0,
    const _Float16* __restrict__ W116, const float* __restrict__ b1,
    const _Float16* __restrict__ W216, const float* __restrict__ b2,
    const float* __restrict__ g_ff, const float* __restrict__ be_ff,
    float* __restrict__ out) {
  __shared__ _Float16 shh[4][16 * 264];  // per-wave 16x256 hidden tile, padded

  const int lane = threadIdx.x & 31;
  const int wid  = threadIdx.x >> 5;
  const int gw   = blockIdx.x * 4 + wid;
  const int b    = gw / (SS / 16);
  const int qb   = (gw % (SS / 16)) * 16;
  const int half = lane >> 4;
  const int nl   = lane & 15;
  _Float16* hbuf = &shh[wid][0];

  const _Float16* arow = H0 + ((size_t)b * SS + qb + nl) * DD;
  v16h af[8];
#pragma unroll
  for (int ks = 0; ks < 8; ++ks)
    af[ks] = ld_pair(arow + ks * 32 + half * 16,
                     arow + ks * 32 + half * 16 + 8);

  // h = relu(H0 @ W1^T + b1), staged to LDS for C->A relayout.
  for (int nt = 0; nt < 16; ++nt) {
    v8f c = vsplat(b1[nt * 16 + nl]);
#pragma unroll
    for (int ks = 0; ks < 8; ++ks)
      c = wmma16(af[ks],
                 ld_row16(W116 + (size_t)(nt * 16 + nl) * DD + ks * 32 + half * 16),
                 c);
#pragma unroll
    for (int r = 0; r < 8; ++r) {
      float v = c[r];
      v = v > 0.f ? v : 0.f;
      hbuf[(half * 8 + r) * 264 + nt * 16 + nl] = (_Float16)v;
    }
  }
  wave_ds_fence();
  v16h hf[8];
#pragma unroll
  for (int ks = 0; ks < 8; ++ks)
    hf[ks] = ld_pair(&hbuf[nl * 264 + ks * 32 + half * 16],
                     &hbuf[nl * 264 + ks * 32 + half * 16 + 8]);

  // y = h @ W2^T + b2
  v8f y[16];
#pragma unroll
  for (int nt = 0; nt < 16; ++nt) {
    v8f c = vsplat(b2[nt * 16 + nl]);
#pragma unroll
    for (int ks = 0; ks < 8; ++ks)
      c = wmma16(hf[ks],
                 ld_row16(W216 + (size_t)(nt * 16 + nl) * DD + ks * 32 + half * 16),
                 c);
    y[nt] = c;
  }

  // final layernorm (no residual, faithful to reference)
  float gv[16], bv[16];
#pragma unroll
  for (int t = 0; t < 16; ++t) { gv[t] = g_ff[t * 16 + nl]; bv[t] = be_ff[t * 16 + nl]; }
  float sum[8], sq[8];
#pragma unroll
  for (int r = 0; r < 8; ++r) { sum[r] = 0.f; sq[r] = 0.f; }
#pragma unroll
  for (int t = 0; t < 16; ++t)
#pragma unroll
    for (int r = 0; r < 8; ++r) { sum[r] += y[t][r]; sq[r] += y[t][r] * y[t][r]; }
#pragma unroll
  for (int r = 0; r < 8; ++r) {
#pragma unroll
    for (int m = 1; m < 16; m <<= 1) {
      sum[r] += __shfl_xor(sum[r], m, 32);
      sq[r]  += __shfl_xor(sq[r],  m, 32);
    }
  }
  float mu[8], rs[8];
#pragma unroll
  for (int r = 0; r < 8; ++r) {
    mu[r] = sum[r] * (1.0f / DD);
    float var = sq[r] * (1.0f / DD) - mu[r] * mu[r];
    rs[r] = rsqrtf(var + LN_EPSF);
  }
#pragma unroll
  for (int t = 0; t < 16; ++t)
#pragma unroll
    for (int r = 0; r < 8; ++r)
      out[((size_t)b * SS + qb + half * 8 + r) * DD + t * 16 + nl] =
          (y[t][r] - mu[r]) * rs[r] * gv[t] + bv[t];
}

// ---------------------------------------------------------------------------
extern "C" void kernel_launch(void* const* d_in, const int* in_sizes, int n_in,
                              void* d_out, int out_size, void* d_ws, size_t ws_size,
                              hipStream_t stream) {
  const float* x     = (const float*)d_in[0];
  const float* Wq    = (const float*)d_in[1];
  const float* bq    = (const float*)d_in[2];
  const float* Wk    = (const float*)d_in[3];
  const float* bk    = (const float*)d_in[4];
  const float* Wv    = (const float*)d_in[5];
  const float* bv    = (const float*)d_in[6];
  const float* W1    = (const float*)d_in[7];
  const float* b1    = (const float*)d_in[8];
  const float* W2    = (const float*)d_in[9];
  const float* b2    = (const float*)d_in[10];
  const float* g_ln  = (const float*)d_in[11];
  const float* be_ln = (const float*)d_in[12];
  const float* g_ff  = (const float*)d_in[13];
  const float* be_ff = (const float*)d_in[14];

  const size_t tok  = (size_t)BB * SS;
  const size_t wsz  = (size_t)DD * DD;     // 65536 elements per weight matrix
  _Float16* Q    = (_Float16*)d_ws;
  _Float16* K    = Q    + tok * DD;        // 8 MB each
  _Float16* Vt   = K    + tok * DD;
  _Float16* OL   = Vt   + tok * DD;        // 32 MB
  _Float16* Wq16 = OL   + tok * DD;        // 128 KB each, L2-resident
  _Float16* Wk16 = Wq16 + wsz;
  _Float16* Wv16 = Wk16 + wsz;
  _Float16* W116 = Wv16 + wsz;
  _Float16* W216 = W116 + wsz;

  // One-shot weight conversions (5 x 64 blocks, negligible).
  dim3 cblk(256), cgrd((unsigned)(wsz / 4 / 256));
  k_cvt4<<<cgrd, cblk, 0, stream>>>((const float4*)Wq, Wq16);
  k_cvt4<<<cgrd, cblk, 0, stream>>>((const float4*)Wk, Wk16);
  k_cvt4<<<cgrd, cblk, 0, stream>>>((const float4*)Wv, Wv16);
  k_cvt4<<<cgrd, cblk, 0, stream>>>((const float4*)W1, W116);
  k_cvt4<<<cgrd, cblk, 0, stream>>>((const float4*)W2, W216);

  dim3 block(128);                       // 4 waves of 32
  dim3 grid((unsigned)(tok / 64));       // 16 rows/wave * 4 waves = 64 rows/WG

  k_qkv<<<grid, block, 0, stream>>>(x, Wq16, bq, Wk16, bk, Wv16, bv, Q, K, Vt);
  k_attn<<<grid, block, 0, stream>>>(Q, K, Vt, x, g_ln, be_ln, OL);
  k_ffn<<<grid, block, 0, stream>>>(OL, W116, b1, W216, b2, g_ff, be_ff, (float*)d_out);
}
